// MolGNN_55181739819227
// MI455X (gfx1250) — matrix operand order repaired and checked by reading, not compiled, not run on hardware
//
#include <hip/hip_runtime.h>
#include <hip/hip_bf16.h>
#include <math.h>

// ---------------- problem constants ----------------
#define NN 100000
#define EE 400000
#define GG 4096
#define HD 512
#define HALFD 256
#define OUTD 768
#define LL 5
#define KA 448   // atom feature K padded (432 -> 448)
#define KE 64    // edge feature K padded (48 -> 64)

typedef __attribute__((ext_vector_type(16))) __bf16 v16bf;
typedef __attribute__((ext_vector_type(8)))  __bf16 v8bf;   // 16 bytes -> global_load_b128
typedef __attribute__((ext_vector_type(8)))  float  v8f;

__device__ __forceinline__ float gelu_f(float x) {
    return 0.5f * x * (1.0f + erff(x * 0.70710678118654752440f));
}
__device__ __forceinline__ unsigned fenc(float f) {
    unsigned u = __float_as_uint(f);
    return (u & 0x80000000u) ? ~u : (u | 0x80000000u);
}
__device__ __forceinline__ float fdec(unsigned k) {
    return (k & 0x80000000u) ? __uint_as_float(k & 0x7fffffffu) : __uint_as_float(~k);
}
__device__ __forceinline__ v16bf cat8(v8bf lo, v8bf hi) {
    return __builtin_shufflevector(lo, hi, 0, 1, 2, 3, 4, 5, 6, 7,
                                   8, 9, 10, 11, 12, 13, 14, 15);
}

// ---------------- utility kernels ----------------
__global__ void k_fill_f32(float* p, float v, long long n) {
    long long i = (long long)blockIdx.x * blockDim.x + threadIdx.x;
    if (i < n) p[i] = v;
}
__global__ void k_init_gmax(unsigned* p, int n) {
    int i = blockIdx.x * blockDim.x + threadIdx.x;
    if (i < n) p[i] = fenc(-3.0e38f);
}
__global__ void k_f32_to_bf16(const float* __restrict__ in, __bf16* __restrict__ out, long long n) {
    long long i = (long long)blockIdx.x * blockDim.x + threadIdx.x;
    if (i < n) out[i] = (__bf16)in[i];
}
// W [K,Nc] row-major f32 -> BT [Nc,Kpad] bf16 (transposed, zero-padded past K)
__global__ void k_cvt_padT(const float* __restrict__ W, __bf16* __restrict__ BT,
                           int K, int Nc, int Kpad) {
    long long i = (long long)blockIdx.x * blockDim.x + threadIdx.x;
    long long tot = (long long)Nc * Kpad;
    if (i >= tot) return;
    int n = (int)(i / Kpad), k = (int)(i % Kpad);
    BT[i] = (k < K) ? (__bf16)W[(long long)k * Nc + n] : (__bf16)0.0f;
}

// ---------------- feature gather kernels ----------------
__global__ void k_atom_feat(const int* __restrict__ x, const float* __restrict__ aemb,
                            __bf16* __restrict__ out) {
    long long i = (long long)blockIdx.x * blockDim.x + threadIdx.x;
    if (i >= (long long)NN * KA) return;
    int n = (int)(i / KA), k = (int)(i % KA);
    float v = 0.0f;
    if (k < 432) {
        int t = k / 48, j = k % 48;
        int idx = x[n * 9 + t];
        v = aemb[((long long)t * 119 + idx) * 48 + j];
    }
    out[i] = (__bf16)v;
}
__global__ void k_edge_feat(const int* __restrict__ eattr, const float* __restrict__ eemb,
                            __bf16* __restrict__ out) {
    long long i = (long long)blockIdx.x * blockDim.x + threadIdx.x;
    if (i >= (long long)EE * KE) return;
    int e = (int)(i >> 6), k = (int)(i & 63);
    float v = 0.0f;
    if (k < 48) {
#pragma unroll
        for (int t = 0; t < 3; ++t) {
            int idx = eattr[e * 3 + t];
            v += eemb[((long long)t * 6 + idx) * 48 + k];
        }
    }
    out[i] = (__bf16)v;
}

// ---------------- WMMA bf16 GEMM ----------------
// C[M,Nc] = act(A[M,K] @ B[K,Nc] + bias[Nc])
// A bf16 row-major [M,K]; BT bf16 row-major [Nc,K] (B transposed). K%32==0, Nc%64==0.
// One wave computes a 32x64 tile: 2 A fragments x 4 B fragments = 8 wmma per k-step.
// Fragment layouts per CDNA5 ISA 7.12.2:
//   A 16x32 bf16: lane(m=l15), lanes0-15 K{0..7,16..23}, lanes16-31 K{8..15,24..31}
//     -> two contiguous 8-elem (16B) runs per lane.
//   B 32x16 bf16: lane(n=l15), lanes0-15 K{0..7,16..23}, lanes16-31 K{8..15,24..31}
//     -> two contiguous 8-elem (16B) runs per lane of BT.
//   C f32 16x16: vgpr r -> row r + 8*half, col l15.
template <int ACT, bool OF32, bool OBF>
__global__ __launch_bounds__(128) void k_gemm_bf16(
    const __bf16* __restrict__ A, const __bf16* __restrict__ BT,
    const float* __restrict__ bias, float* __restrict__ Cf, __bf16* __restrict__ Cb,
    int M, int Nc, int K)
{
    const int lane   = threadIdx.x & 31;
    const int wave   = threadIdx.x >> 5;
    const int tileM0 = (blockIdx.y * 4 + wave) * 32;  // 32 rows per wave
    const int nBase  = blockIdx.x * 64;               // 64 cols per wave
    const int half   = lane >> 4;
    const int l15    = lane & 15;

    int m0 = tileM0 + l15;
    int m1 = tileM0 + 16 + l15;
    int m0c = (m0 < M) ? m0 : (M - 1);
    int m1c = (m1 < M) ? m1 : (M - 1);
    const __bf16* arow0 = A + (long long)m0c * K + half * 8;
    const __bf16* arow1 = A + (long long)m1c * K + half * 8;

    v8f acc[2][4];
#pragma unroll
    for (int g = 0; g < 2; ++g)
#pragma unroll
        for (int t = 0; t < 4; ++t)
#pragma unroll
            for (int i = 0; i < 8; ++i) acc[g][t][i] = 0.0f;

    for (int kb = 0; kb < K; kb += 32) {
        v16bf a0 = cat8(*(const v8bf*)(arow0 + kb), *(const v8bf*)(arow0 + kb + 16));
        v16bf a1 = cat8(*(const v8bf*)(arow1 + kb), *(const v8bf*)(arow1 + kb + 16));
#pragma unroll
        for (int t = 0; t < 4; ++t) {
            int n = nBase + t * 16 + l15;
            const __bf16* bp = BT + (long long)n * K + kb + half * 8;
            v16bf bv = cat8(*(const v8bf*)bp, *(const v8bf*)(bp + 16));
            acc[0][t] = __builtin_amdgcn_wmma_f32_16x16x32_bf16(
                false, a0, false, bv, (short)0, acc[0][t], false, false);
            acc[1][t] = __builtin_amdgcn_wmma_f32_16x16x32_bf16(
                false, a1, false, bv, (short)0, acc[1][t], false, false);
        }
    }

#pragma unroll
    for (int g = 0; g < 2; ++g) {
        int rowBase = tileM0 + g * 16 + half * 8;
#pragma unroll
        for (int t = 0; t < 4; ++t) {
            int col = nBase + t * 16 + l15;
            float bs = bias[col];
#pragma unroll
            for (int r = 0; r < 8; ++r) {
                int row = rowBase + r;
                if (row < M) {
                    float v = acc[g][t][r] + bs;
                    if (ACT == 1) v = gelu_f(v);
                    long long o = (long long)row * Nc + col;
                    if constexpr (OF32) Cf[o] = v;
                    if constexpr (OBF)  Cb[o] = (__bf16)v;
                }
            }
        }
    }
}

// ---------------- GINE layer kernels ----------------
// one block per edge; each thread handles 2 consecutive channels (256 thr x 2 = 512)
__global__ void k_message(const float* __restrict__ h, const __bf16* __restrict__ e,
                          const int* __restrict__ src, const int* __restrict__ dst,
                          float* __restrict__ agg) {
    int eidx = blockIdx.x;
    int s = src[eidx], d = dst[eidx];
    int c = threadIdx.x;
    const float2* hr = (const float2*)(h + (long long)s * HD);
    const unsigned* er = (const unsigned*)(e + (long long)eidx * HD);
    float* ar = agg + (long long)d * HD;
    float2 hv = hr[c];
    unsigned eu = er[c];
    float e0 = __uint_as_float(eu << 16);           // bf16 lo -> f32
    float e1 = __uint_as_float(eu & 0xffff0000u);   // bf16 hi -> f32
    float m0 = fmaxf(hv.x + e0, 0.0f);
    float m1 = fmaxf(hv.y + e1, 0.0f);
    atomicAdd(&ar[2 * c], m0);
    atomicAdd(&ar[2 * c + 1], m1);
}
__global__ void k_zin(const float* __restrict__ h, const float* __restrict__ agg,
                      const float* __restrict__ eps, int l, __bf16* __restrict__ zin) {
    long long i = (long long)blockIdx.x * blockDim.x + threadIdx.x;
    if (i < (long long)NN * HD) zin[i] = (__bf16)((1.0f + eps[l]) * h[i] + agg[i]);
}
__global__ void k_ln_residual(float* __restrict__ h, const float* __restrict__ z2,
                              const float* __restrict__ lg, const float* __restrict__ lb) {
    __shared__ float red[256];
    int n = blockIdx.x, tid = threadIdx.x;
    const float* zr = z2 + (long long)n * HD;
    float s = 0.0f;
    for (int c = tid; c < HD; c += 256) s += zr[c];
    red[tid] = s; __syncthreads();
    for (int st = 128; st > 0; st >>= 1) { if (tid < st) red[tid] += red[tid + st]; __syncthreads(); }
    float mean = red[0] / (float)HD; __syncthreads();
    float s2 = 0.0f;
    for (int c = tid; c < HD; c += 256) { float d = zr[c] - mean; s2 += d * d; }
    red[tid] = s2; __syncthreads();
    for (int st = 128; st > 0; st >>= 1) { if (tid < st) red[tid] += red[tid + st]; __syncthreads(); }
    float rstd = rsqrtf(red[0] / (float)HD + 1e-5f);
    float* hr = h + (long long)n * HD;
    for (int c = tid; c < HD; c += 256) {
        float nv = (zr[c] - mean) * rstd * lg[c] + lb[c];
        hr[c] = hr[c] + gelu_f(nv);
    }
}

// ---------------- attention aggregation ----------------
__global__ void k_gate_dot(const __bf16* __restrict__ gt, const float* __restrict__ w2,
                           const float* __restrict__ b2, float* __restrict__ gate) {
    __shared__ float red[256];
    int n = blockIdx.x, tid = threadIdx.x;
    float v = (float)gt[(long long)n * HALFD + tid] * w2[tid];
    red[tid] = v; __syncthreads();
    for (int st = 128; st > 0; st >>= 1) { if (tid < st) red[tid] += red[tid + st]; __syncthreads(); }
    if (tid == 0) gate[n] = red[0] + b2[0];
}
__global__ void k_segmax(const float* __restrict__ gate, const int* __restrict__ batch,
                         unsigned* __restrict__ gmax) {
    int n = blockIdx.x * blockDim.x + threadIdx.x;
    if (n < NN) atomicMax(&gmax[batch[n]], fenc(gate[n]));
}
__global__ void k_expsum(const float* __restrict__ gate, const int* __restrict__ batch,
                         const unsigned* __restrict__ gmax, float* __restrict__ a,
                         float* __restrict__ denom) {
    int n = blockIdx.x * blockDim.x + threadIdx.x;
    if (n < NN) {
        int b = batch[n];
        float v = expf(gate[n] - fdec(gmax[b]));
        a[n] = v;
        atomicAdd(&denom[b], v);
    }
}
__global__ void k_attn_wsum(const float* __restrict__ h, const float* __restrict__ a,
                            const int* __restrict__ batch, float* __restrict__ gsum) {
    int n = blockIdx.x;
    float an = a[n];
    int b = batch[n];
    const float* hr = h + (long long)n * HD;
    float* gr = gsum + (long long)b * HD;
    for (int c = threadIdx.x; c < HD; c += blockDim.x) atomicAdd(&gr[c], an * hr[c]);
}
__global__ void k_gdiv(const float* __restrict__ gsum, const float* __restrict__ denom,
                       __bf16* __restrict__ gbf) {
    long long i = (long long)blockIdx.x * blockDim.x + threadIdx.x;
    if (i < (long long)GG * HD) gbf[i] = (__bf16)(gsum[i] / denom[i >> 9]);
}
__global__ void k_outnorm(const float* __restrict__ g3, float* __restrict__ out) {
    __shared__ float red[256];
    int g = blockIdx.x, tid = threadIdx.x;
    const float* r = g3 + (long long)g * OUTD;
    float s = 0.0f;
    for (int c = tid; c < OUTD; c += 256) s += r[c] * r[c];
    red[tid] = s; __syncthreads();
    for (int st = 128; st > 0; st >>= 1) { if (tid < st) red[tid] += red[tid + st]; __syncthreads(); }
    float inv = 1.0f / (sqrtf(red[0]) + 1e-12f);
    for (int c = tid; c < OUTD; c += 256) out[(long long)g * OUTD + c] = r[c] * inv;
}

// ---------------- host side ----------------
static inline dim3 gemm_grid(int M, int Nc) { return dim3(Nc / 64, (M + 127) / 128); }
static inline int ceil_div_ll(long long n, int b) { return (int)((n + b - 1) / b); }

extern "C" void kernel_launch(void* const* d_in, const int* in_sizes, int n_in,
                              void* d_out, int out_size, void* d_ws, size_t ws_size,
                              hipStream_t stream) {
    (void)in_sizes; (void)n_in; (void)out_size; (void)ws_size;
    const int*   x         = (const int*)d_in[0];
    const int*   edge_attr = (const int*)d_in[1];
    const int*   edge_idx  = (const int*)d_in[2];
    const int*   batch     = (const int*)d_in[3];
    const float* atom_emb  = (const float*)d_in[4];
    const float* atomW     = (const float*)d_in[5];
    const float* atomB     = (const float*)d_in[6];
    const float* edge_emb  = (const float*)d_in[7];
    const float* edgeW     = (const float*)d_in[8];
    const float* edgeB     = (const float*)d_in[9];
    const float* convW1    = (const float*)d_in[10];
    const float* convb1    = (const float*)d_in[11];
    const float* convW2    = (const float*)d_in[12];
    const float* convb2    = (const float*)d_in[13];
    const float* eps       = (const float*)d_in[14];
    const float* ln_g      = (const float*)d_in[15];
    const float* ln_b      = (const float*)d_in[16];
    const float* gateW1    = (const float*)d_in[17];
    const float* gateb1    = (const float*)d_in[18];
    const float* gateW2    = (const float*)d_in[19];
    const float* gateb2    = (const float*)d_in[20];
    const float* roW       = (const float*)d_in[21];
    const float* rob       = (const float*)d_in[22];
    const float* pW1       = (const float*)d_in[23];
    const float* pb1       = (const float*)d_in[24];
    const float* pW2       = (const float*)d_in[25];
    const float* pb2       = (const float*)d_in[26];
    float* out = (float*)d_out;

    const int* src = edge_idx;
    const int* dst = edge_idx + EE;

    char* ws = (char*)d_ws;
    size_t off = 0;
    auto alloc = [&](size_t bytes) -> void* {
        off = (off + 255) & ~(size_t)255;
        void* p = ws + off;
        off += bytes;
        return p;
    };
    // transposed bf16 weights: BT[Nc, Kpad]
    __bf16* wAtom  = (__bf16*)alloc((size_t)HD * KA * 2);
    __bf16* wEdge  = (__bf16*)alloc((size_t)HD * KE * 2);
    __bf16* wC1    = (__bf16*)alloc((size_t)LL * HD * HD * 2);
    __bf16* wC2    = (__bf16*)alloc((size_t)LL * HD * HD * 2);
    __bf16* wG1    = (__bf16*)alloc((size_t)HALFD * HD * 2);
    __bf16* wRo    = (__bf16*)alloc((size_t)HD * HD * 2);
    __bf16* wP1    = (__bf16*)alloc((size_t)HD * HD * 2);
    __bf16* wP2    = (__bf16*)alloc((size_t)OUTD * HD * 2);
    __bf16* atomA  = (__bf16*)alloc((size_t)NN * KA * 2);
    __bf16* edgeA  = (__bf16*)alloc((size_t)EE * KE * 2);
    float*  h      = (float*) alloc((size_t)NN * HD * 4);
    __bf16* e_bf   = (__bf16*)alloc((size_t)EE * HD * 2);
    float*  agg    = (float*) alloc((size_t)NN * HD * 4);   // reused as z2
    __bf16* zin    = (__bf16*)alloc((size_t)NN * HD * 2);   // reused as h_bf
    __bf16* t1     = (__bf16*)alloc((size_t)NN * HD * 2);   // reused as gate hidden
    float*  gate   = (float*) alloc((size_t)NN * 4);
    float*  aexp   = (float*) alloc((size_t)NN * 4);
    unsigned* gmax = (unsigned*)alloc((size_t)GG * 4);
    float*  denom  = (float*) alloc((size_t)GG * 4);
    float*  gsum   = (float*) alloc((size_t)GG * HD * 4);
    __bf16* g_bf   = (__bf16*)alloc((size_t)GG * HD * 2);
    __bf16* g1_bf  = (__bf16*)alloc((size_t)GG * HD * 2);
    __bf16* g2_bf  = (__bf16*)alloc((size_t)GG * HD * 2);
    float*  g3     = (float*) alloc((size_t)GG * OUTD * 4);
    float*  z2     = agg;
    __bf16* h_bf   = zin;
    __bf16* gt_bf  = t1;

    const dim3 B256(256);

    // ---- weight conversion (transpose + pad to bf16) ----
    k_cvt_padT<<<ceil_div_ll((long long)HD * KA, 256), B256, 0, stream>>>(atomW, wAtom, 432, HD, KA);
    k_cvt_padT<<<ceil_div_ll((long long)HD * KE, 256), B256, 0, stream>>>(edgeW, wEdge, 48, HD, KE);
    for (int l = 0; l < LL; ++l) {
        k_cvt_padT<<<ceil_div_ll((long long)HD * HD, 256), B256, 0, stream>>>(
            convW1 + (size_t)l * HD * HD, wC1 + (size_t)l * HD * HD, HD, HD, HD);
        k_cvt_padT<<<ceil_div_ll((long long)HD * HD, 256), B256, 0, stream>>>(
            convW2 + (size_t)l * HD * HD, wC2 + (size_t)l * HD * HD, HD, HD, HD);
    }
    k_cvt_padT<<<ceil_div_ll((long long)HALFD * HD, 256), B256, 0, stream>>>(gateW1, wG1, HD, HALFD, HD);
    k_cvt_padT<<<ceil_div_ll((long long)HD * HD, 256), B256, 0, stream>>>(roW, wRo, HD, HD, HD);
    k_cvt_padT<<<ceil_div_ll((long long)HD * HD, 256), B256, 0, stream>>>(pW1, wP1, HD, HD, HD);
    k_cvt_padT<<<ceil_div_ll((long long)OUTD * HD, 256), B256, 0, stream>>>(pW2, wP2, HD, OUTD, HD);

    // ---- encoders ----
    k_atom_feat<<<ceil_div_ll((long long)NN * KA, 256), B256, 0, stream>>>(x, atom_emb, atomA);
    k_edge_feat<<<ceil_div_ll((long long)EE * KE, 256), B256, 0, stream>>>(edge_attr, edge_emb, edgeA);
    k_gemm_bf16<1, true, false><<<gemm_grid(NN, HD), dim3(128), 0, stream>>>(
        atomA, wAtom, atomB, h, nullptr, NN, HD, KA);
    k_gemm_bf16<1, false, true><<<gemm_grid(EE, HD), dim3(128), 0, stream>>>(
        edgeA, wEdge, edgeB, nullptr, e_bf, EE, HD, KE);

    // ---- GINE conv stack ----
    for (int l = 0; l < LL; ++l) {
        k_fill_f32<<<ceil_div_ll((long long)NN * HD, 256), B256, 0, stream>>>(agg, 0.0f, (long long)NN * HD);
        k_message<<<dim3(EE), B256, 0, stream>>>(h, e_bf, src, dst, agg);
        k_zin<<<ceil_div_ll((long long)NN * HD, 256), B256, 0, stream>>>(h, agg, eps, l, zin);
        k_gemm_bf16<1, false, true><<<gemm_grid(NN, HD), dim3(128), 0, stream>>>(
            zin, wC1 + (size_t)l * HD * HD, convb1 + (size_t)l * HD, nullptr, t1, NN, HD, HD);
        k_gemm_bf16<0, true, false><<<gemm_grid(NN, HD), dim3(128), 0, stream>>>(
            t1, wC2 + (size_t)l * HD * HD, convb2 + (size_t)l * HD, z2, nullptr, NN, HD, HD);
        k_ln_residual<<<dim3(NN), B256, 0, stream>>>(h, z2, ln_g + (size_t)l * HD, ln_b + (size_t)l * HD);
    }

    // ---- attentional aggregation ----
    k_f32_to_bf16<<<ceil_div_ll((long long)NN * HD, 256), B256, 0, stream>>>(h, h_bf, (long long)NN * HD);
    k_gemm_bf16<1, false, true><<<gemm_grid(NN, HALFD), dim3(128), 0, stream>>>(
        h_bf, wG1, gateb1, nullptr, gt_bf, NN, HALFD, HD);
    k_gate_dot<<<dim3(NN), B256, 0, stream>>>(gt_bf, gateW2, gateb2, gate);
    k_init_gmax<<<ceil_div_ll(GG, 256), B256, 0, stream>>>(gmax, GG);
    k_fill_f32<<<ceil_div_ll(GG, 256), B256, 0, stream>>>(denom, 0.0f, GG);
    k_fill_f32<<<ceil_div_ll((long long)GG * HD, 256), B256, 0, stream>>>(gsum, 0.0f, (long long)GG * HD);
    k_segmax<<<ceil_div_ll(NN, 256), B256, 0, stream>>>(gate, batch, gmax);
    k_expsum<<<ceil_div_ll(NN, 256), B256, 0, stream>>>(gate, batch, gmax, aexp, denom);
    k_attn_wsum<<<dim3(NN), B256, 0, stream>>>(h, aexp, batch, gsum);
    k_gdiv<<<ceil_div_ll((long long)GG * HD, 256), B256, 0, stream>>>(gsum, denom, g_bf);

    // ---- readout ----
    k_gemm_bf16<1, false, true><<<gemm_grid(GG, HD), dim3(128), 0, stream>>>(
        g_bf, wRo, rob, nullptr, g1_bf, GG, HD, HD);
    k_gemm_bf16<1, false, true><<<gemm_grid(GG, HD), dim3(128), 0, stream>>>(
        g1_bf, wP1, pb1, nullptr, g2_bf, GG, HD, HD);
    k_gemm_bf16<0, true, false><<<gemm_grid(GG, OUTD), dim3(128), 0, stream>>>(
        g2_bf, wP2, pb2, g3, nullptr, GG, OUTD, HD);
    k_outnorm<<<dim3(GG), B256, 0, stream>>>(g3, out);
}
